// SMILESDecoder_7825430413586
// MI455X (gfx1250) — matrix operand order repaired and checked by reading, not compiled, not run on hardware
//
#include <hip/hip_runtime.h>

// ---------------------------------------------------------------------------
// SMILES transformer decoder for MI455X (gfx1250), wave32 + WMMA bf16.
// - All GEMMs via v_wmma_f32_16x16x32_bf16 (fp32 accumulate).
// - bf16 tiles staged with global_load_async_to_lds_b128 (ASYNCcnt path).
// - Streamed fp32 weights prefetched with global_prefetch_b8.
// - Cross-attention exact-simplified (uniform softmax over identical
//   broadcast keys) to two tiny GEMMs.
// ---------------------------------------------------------------------------

typedef __attribute__((ext_vector_type(16))) __bf16 v16bf;
typedef __attribute__((ext_vector_type(8)))  float  v8f;

union BF16x16 { v16bf v; uint4 u[2]; };

static constexpr int Bc  = 16;
static constexpr int Sc  = 512;
static constexpr int Dc  = 768;
static constexpr int Hc  = 8;
static constexpr int DHc = 96;
static constexpr int Fc  = 2048;
static constexpr int Vc  = 100;
static constexpr int Lc  = 6;
static constexpr int MR  = Bc * Sc;        // 8192 rows

__device__ __forceinline__ unsigned short f2bf(float f) {
    unsigned int u = __float_as_uint(f);
    u += 0x7FFFu + ((u >> 16) & 1u);       // round-to-nearest-even
    return (unsigned short)(u >> 16);
}
__device__ __forceinline__ unsigned int pk2(float a, float b) {
    return (unsigned int)f2bf(a) | ((unsigned int)f2bf(b) << 16);
}
__device__ __forceinline__ v8f wmma_bf16(const v16bf& a, const v16bf& b, const v8f& c) {
    return __builtin_amdgcn_wmma_f32_16x16x32_bf16(false, a, false, b, (short)0, c, false, false);
}
// Flat shared pointer -> 32-bit LDS byte address (aperture low bits, ISA §10.2).
__device__ __forceinline__ unsigned lds_addr32(const void* p) {
    return (unsigned)(unsigned long long)p;
}
// Async DMA: 32 contiguous bytes global -> LDS, no VGPR round trip.
// INST_OFFSET is added to BOTH global and LDS addresses (ISA §10.7/15.18).
__device__ __forceinline__ void async_copy_b32bytes(unsigned lds, const void* gptr) {
    asm volatile("global_load_async_to_lds_b128 %0, %1, off\n\t"
                 "global_load_async_to_lds_b128 %0, %1, off offset:16"
                 :: "v"(lds), "v"(gptr) : "memory");
}
__device__ __forceinline__ void wait_async0() {
    asm volatile("s_wait_asynccnt 0x0" ::: "memory");
}

// ---------------------------------------------------------------------------
// Embedding: x[row,:] = tok_emb[target[row]] + pos_emb[row % S]  (f32 + bf16)
// ---------------------------------------------------------------------------
__global__ __launch_bounds__(256) void k_embed(const int* __restrict__ target,
                                               const float* __restrict__ tok,
                                               const float* __restrict__ pos,
                                               float* __restrict__ xf,
                                               unsigned short* __restrict__ xb) {
    int row = blockIdx.x;
    int s   = row & (Sc - 1);
    int t   = target[row];
    int tid = threadIdx.x;
#pragma unroll
    for (int i = 0; i < 3; ++i) {
        int c = tid + i * 256;
        float f = tok[(size_t)t * Dc + c] + pos[(size_t)s * Dc + c];
        xf[(size_t)row * Dc + c] = f;
        xb[(size_t)row * Dc + c] = f2bf(f);
    }
}

__global__ __launch_bounds__(256) void k_cvt(const float* __restrict__ in,
                                             unsigned short* __restrict__ out, int n) {
    int i = blockIdx.x * 256 + threadIdx.x;
    if (i < n) out[i] = f2bf(in[i]);
}

// ---------------------------------------------------------------------------
// GEMM: out[M,N] = act(A_bf16[M,K] @ W_f32[N,K]^T + bias)  -> f32 and/or bf16
// Block tile 128x128, BK=32, 8 waves (2x4), wave tile 64x32 (4x2 WMMA tiles).
// A tile: async DMA to LDS.  W tile: f32->bf16 through VGPRs + prefetch.
// ---------------------------------------------------------------------------
__global__ __launch_bounds__(256) void k_gemm(const unsigned short* __restrict__ A,
                                              const float* __restrict__ W,
                                              const float* __restrict__ bias,
                                              float* __restrict__ outF,
                                              unsigned short* __restrict__ outB,
                                              int M, int N, int K, int relu) {
    __shared__ alignas(16) unsigned short As[128 * 40];  // pad 32->40 halves
    __shared__ alignas(16) unsigned short Bs[128 * 40];

    const int tid  = threadIdx.x;
    const int lane = tid & 31;
    const int wave = tid >> 5;
    const int wr   = wave >> 2;           // 0..1 -> 64-row slab
    const int wc   = wave & 3;            // 0..3 -> 32-col slab
    const int bm   = blockIdx.y * 128;
    const int bn   = blockIdx.x * 128;

    const int rl   = lane & 15;
    const int hi   = lane >> 4;

    const v8f vzero = {0.f, 0.f, 0.f, 0.f, 0.f, 0.f, 0.f, 0.f};
    v8f acc[4][2];
#pragma unroll
    for (int i = 0; i < 4; ++i)
#pragma unroll
        for (int j = 0; j < 2; ++j) acc[i][j] = vzero;

    const int lrow = tid >> 1;            // 0..127
    const int lkh  = (tid & 1) * 16;      // 0 / 16
    const unsigned ldsA = lds_addr32(As + lrow * 40 + lkh);

    for (int k0 = 0; k0 < K; k0 += 32) {
        // --- stage A tile: async global->LDS (bf16 pass-through) ---
        {
            int gr = bm + lrow;
            if (gr < M) {
                async_copy_b32bytes(ldsA, A + (size_t)gr * K + k0 + lkh);
            } else {
                uint4 z{};
                uint4* q = (uint4*)(As + lrow * 40 + lkh);
                q[0] = z; q[1] = z;
            }
        }
        // --- stage W tile (f32 -> bf16 conversion in VGPRs) ---
        {
            float4 f0{}, f1{}, f2{}, f3{};
            int gn = bn + lrow;
            if (gn < N) {
                const float4* p = (const float4*)(W + (size_t)gn * K + k0 + lkh);
                f0 = p[0]; f1 = p[1]; f2 = p[2]; f3 = p[3];
                if (k0 + 32 < K)  // stream-prefetch next K slab of W
                    __builtin_prefetch(W + (size_t)gn * K + k0 + 32 + lkh, 0, 1);
            }
            uint4 w0, w1;
            w0.x = pk2(f0.x, f0.y); w0.y = pk2(f0.z, f0.w);
            w0.z = pk2(f1.x, f1.y); w0.w = pk2(f1.z, f1.w);
            w1.x = pk2(f2.x, f2.y); w1.y = pk2(f2.z, f2.w);
            w1.z = pk2(f3.x, f3.y); w1.w = pk2(f3.z, f3.w);
            uint4* q = (uint4*)(Bs + lrow * 40 + lkh);
            q[0] = w0; q[1] = w1;
        }
        wait_async0();                    // our async DMA done before barrier
        __syncthreads();

        // --- fragments + 8 WMMAs ---
        const int kb = hi ? 16 : 0;       // A: K range per half-wave
        const int ks = hi ? 16 : 0;       // B: K range per half-wave
        BF16x16 afr[4], bfr[2];
#pragma unroll
        for (int mt = 0; mt < 4; ++mt) {
            const uint4* p = (const uint4*)(As + (wr * 64 + mt * 16 + rl) * 40 + kb);
            afr[mt].u[0] = p[0];          // K = kb .. kb+7
            afr[mt].u[1] = p[1];          // K = kb+8 .. kb+15
        }
#pragma unroll
        for (int nt = 0; nt < 2; ++nt) {
            const uint4* p = (const uint4*)(Bs + (wc * 32 + nt * 16 + rl) * 40 + ks);
            bfr[nt].u[0] = p[0];          // K = ks .. ks+15 contiguous
            bfr[nt].u[1] = p[1];
        }
#pragma unroll
        for (int mt = 0; mt < 4; ++mt)
#pragma unroll
            for (int nt = 0; nt < 2; ++nt)
                acc[mt][nt] = wmma_bf16(afr[mt].v, bfr[nt].v, acc[mt][nt]);
        __syncthreads();
    }

    // --- epilogue: C layout N=lane&15, M=v+8*hi ---
#pragma unroll
    for (int mt = 0; mt < 4; ++mt) {
#pragma unroll
        for (int nt = 0; nt < 2; ++nt) {
            int col = bn + wc * 32 + nt * 16 + rl;
            if (col >= N) continue;
            float bv = bias ? bias[col] : 0.f;
#pragma unroll
            for (int v = 0; v < 8; ++v) {
                int row = bm + wr * 64 + mt * 16 + hi * 8 + v;
                if (row >= M) continue;
                float val = acc[mt][nt][v] + bv;
                if (relu) val = fmaxf(val, 0.f);
                size_t idx = (size_t)row * N + col;
                if (outF) outF[idx] = val;
                if (outB) outB[idx] = f2bf(val);
            }
        }
    }
}

// ---------------------------------------------------------------------------
// Causal self-attention (flash style). qkv bf16 [MR, 3D]; out bf16 [MR, D].
// Block = (b, h, 128 q-rows); 8 waves, each owns a 16-row q tile.
// K tile staged via async DMA; V staged transposed (scalar path).
// Inner step = 32 keys: QK^T via 6 WMMAs, online softmax, P@V via 6 WMMAs.
// ---------------------------------------------------------------------------
__global__ __launch_bounds__(256) void k_attn(const unsigned short* __restrict__ qkv,
                                              unsigned short* __restrict__ out) {
    __shared__ alignas(16) unsigned short Kt[32 * 104];     // [key][dh], pad
    __shared__ alignas(16) unsigned short Vt[96 * 40];      // [dh][key], pad
    __shared__ alignas(16) unsigned short Pt[8 * 16 * 40];  // per-wave P tile

    const int tid  = threadIdx.x;
    const int lane = tid & 31;
    const int wave = tid >> 5;
    const int rl   = lane & 15;
    const int hi   = lane >> 4;

    const int b   = blockIdx.z;
    const int h   = blockIdx.y;
    const int qb0 = blockIdx.x * 128;
    const int qbase = qb0 + wave * 16;

    const float scale = 0.1020620726f;    // 1/sqrt(96)

    // Q fragments (K = 96 -> 3 chunks of 32)
    BF16x16 qf[3];
    {
        const unsigned short* qrow =
            qkv + (size_t)(b * Sc + qbase + rl) * (3 * Dc) + h * DHc;
        const int kb = hi ? 16 : 0;
#pragma unroll
        for (int c = 0; c < 3; ++c) {
            const uint4* p = (const uint4*)(qrow + c * 32 + kb);
            qf[c].u[0] = p[0];
            qf[c].u[1] = p[1];
        }
    }

    const v8f vzero = {0.f, 0.f, 0.f, 0.f, 0.f, 0.f, 0.f, 0.f};
    v8f o[6];
#pragma unroll
    for (int c = 0; c < 6; ++c) o[c] = vzero;
    float rmax[8], rsum[8];
#pragma unroll
    for (int v = 0; v < 8; ++v) { rmax[v] = -1e30f; rsum[v] = 0.f; }

    // K-tile DMA assignment: threads 0..191 (exactly 6 waves, no divergence)
    const int kkey = tid / 6;             // 0..31 (valid for tid<192)
    const int kch  = tid - kkey * 6;      // 0..5  (16 dh cols = 32B each)
    const unsigned ldsK = lds_addr32(Kt + kkey * 104 + kch * 16);

    const int nj = (qb0 + 128) / 32;      // causal bound for this block

    for (int j = 0; j < nj; ++j) {
        const int kk0 = j * 32;
        // --- stage K via async DMA (bf16 pass-through) ---
        if (tid < 192) {
            const unsigned short* gk =
                qkv + (size_t)(b * Sc + kk0 + kkey) * (3 * Dc) + Dc + h * DHc + kch * 16;
            async_copy_b32bytes(ldsK, gk);
        }
        // --- stage V transposed (scalar; transpose can't DMA) ---
        for (int idx = tid; idx < 32 * 96; idx += 256) {
            int r  = idx / 96;
            int c2 = idx - r * 96;
            Vt[c2 * 40 + r] =
                qkv[(size_t)(b * Sc + kk0 + r) * (3 * Dc) + 2 * Dc + h * DHc + c2];
        }
        wait_async0();
        __syncthreads();

        if (kk0 <= qbase + 15) {          // wave-uniform: any unmasked key?
            const int ds0 = hi ? 16 : 0;
            v8f s0 = vzero, s1 = vzero;
#pragma unroll
            for (int c = 0; c < 3; ++c) {
                BF16x16 b0, b1;
                const uint4* p0 = (const uint4*)(Kt + rl * 104 + c * 32 + ds0);
                b0.u[0] = p0[0]; b0.u[1] = p0[1];
                const uint4* p1 = (const uint4*)(Kt + (16 + rl) * 104 + c * 32 + ds0);
                b1.u[0] = p1[0]; b1.u[1] = p1[1];
                s0 = wmma_bf16(qf[c].v, b0.v, s0);
                s1 = wmma_bf16(qf[c].v, b1.v, s1);
            }

            float corr[8];
#pragma unroll
            for (int v = 0; v < 8; ++v) {
                int m    = qbase + hi * 8 + v;
                int key0 = kk0 + rl;
                float sv0 = (key0      > m) ? -1e30f : s0[v] * scale;
                float sv1 = (key0 + 16 > m) ? -1e30f : s1[v] * scale;
                float t = fmaxf(sv0, sv1);
#pragma unroll
                for (int off = 1; off < 16; off <<= 1)
                    t = fmaxf(t, __shfl_xor(t, off, 16));
                float nm = fmaxf(rmax[v], t);
                float cr = __expf(rmax[v] - nm);
                float p0 = __expf(sv0 - nm);
                float p1 = __expf(sv1 - nm);
                float ps = p0 + p1;
#pragma unroll
                for (int off = 1; off < 16; off <<= 1)
                    ps += __shfl_xor(ps, off, 16);
                rsum[v] = rsum[v] * cr + ps;
                rmax[v] = nm;
                corr[v] = cr;
                int prow = hi * 8 + v;
                Pt[wave * 640 + prow * 40 + rl]      = f2bf(p0);
                Pt[wave * 640 + prow * 40 + 16 + rl] = f2bf(p1);
            }
#pragma unroll
            for (int c2 = 0; c2 < 6; ++c2)
#pragma unroll
                for (int v = 0; v < 8; ++v) o[c2][v] *= corr[v];

            // wave-local LDS RAW: P written cross-lane, read as A fragment
            asm volatile("s_wait_dscnt 0" ::: "memory");

            BF16x16 pa;
            {
                const int kb = hi ? 16 : 0;
                const uint4* p = (const uint4*)(Pt + wave * 640 + rl * 40 + kb);
                pa.u[0] = p[0];
                pa.u[1] = p[1];
            }
#pragma unroll
            for (int c2 = 0; c2 < 6; ++c2) {
                BF16x16 vb;
                const uint4* p = (const uint4*)(Vt + (c2 * 16 + rl) * 40 + ds0);
                vb.u[0] = p[0]; vb.u[1] = p[1];
                o[c2] = wmma_bf16(pa.v, vb.v, o[c2]);
            }
        }
        __syncthreads();
    }

    // normalize + store bf16
#pragma unroll
    for (int v = 0; v < 8; ++v) {
        float inv = 1.f / rsum[v];
        int m = qbase + hi * 8 + v;
        unsigned short* orow = out + (size_t)(b * Sc + m) * Dc + h * DHc;
#pragma unroll
        for (int c2 = 0; c2 < 6; ++c2)
            orow[c2 * 16 + rl] = f2bf(o[c2][v] * inv);
    }
}

// ---------------------------------------------------------------------------
// Fused residual add + LayerNorm. One block per row (D=768, 256 threads).
// bcast=1: y indexed by batch (cross-attn broadcast add).
// ---------------------------------------------------------------------------
__global__ __launch_bounds__(256) void k_addln(const float* __restrict__ x,
                                               const float* __restrict__ y,
                                               const float* __restrict__ g,
                                               const float* __restrict__ be,
                                               float* __restrict__ xf,
                                               unsigned short* __restrict__ xb,
                                               int bcast) {
    __shared__ float r1[256], r2[256];
    const int row  = blockIdx.x;
    const int tid  = threadIdx.x;
    const int yrow = bcast ? (row >> 9) : row;   // row / S

    float v[3];
    float s1 = 0.f, s2 = 0.f;
#pragma unroll
    for (int i = 0; i < 3; ++i) {
        int c = tid + i * 256;
        float f = x[(size_t)row * Dc + c] + y[(size_t)yrow * Dc + c];
        v[i] = f; s1 += f; s2 += f * f;
    }
    r1[tid] = s1; r2[tid] = s2;
    __syncthreads();
    for (int st = 128; st > 0; st >>= 1) {
        if (tid < st) { r1[tid] += r1[tid + st]; r2[tid] += r2[tid + st]; }
        __syncthreads();
    }
    float mu   = r1[0] * (1.f / Dc);
    float var  = r2[0] * (1.f / Dc) - mu * mu;
    float rstd = rsqrtf(var + 1e-5f);
#pragma unroll
    for (int i = 0; i < 3; ++i) {
        int c = tid + i * 256;
        float o = (v[i] - mu) * rstd * g[c] + be[c];
        xf[(size_t)row * Dc + c] = o;
        xb[(size_t)row * Dc + c] = f2bf(o);
    }
}

// ---------------------------------------------------------------------------
// Host orchestration
// ---------------------------------------------------------------------------
extern "C" void kernel_launch(void* const* d_in, const int* in_sizes, int n_in,
                              void* d_out, int out_size, void* d_ws, size_t ws_size,
                              hipStream_t stream) {
    (void)in_sizes; (void)n_in; (void)out_size; (void)ws_size;

    const float* latent  = (const float*)d_in[0];
    const int*   target  = (const int*)d_in[1];
    const float* tok_emb = (const float*)d_in[2];
    const float* pos_emb = (const float*)d_in[3];
    const float* sa_w_qkv = (const float*)d_in[4];
    const float* sa_b_qkv = (const float*)d_in[5];
    const float* sa_w_o   = (const float*)d_in[6];
    const float* sa_b_o   = (const float*)d_in[7];
    const float* ca_w_qkv = (const float*)d_in[8];
    const float* ca_b_qkv = (const float*)d_in[9];
    const float* ca_w_o   = (const float*)d_in[10];
    const float* ca_b_o   = (const float*)d_in[11];
    const float* ffn_w1   = (const float*)d_in[12];
    const float* ffn_b1   = (const float*)d_in[13];
    const float* ffn_w2   = (const float*)d_in[14];
    const float* ffn_b2   = (const float*)d_in[15];
    const float* ln1_g = (const float*)d_in[16];
    const float* ln1_b = (const float*)d_in[17];
    const float* ln2_g = (const float*)d_in[18];
    const float* ln2_b = (const float*)d_in[19];
    const float* ln3_g = (const float*)d_in[20];
    const float* ln3_b = (const float*)d_in[21];
    const float* out_w = (const float*)d_in[22];
    const float* out_b = (const float*)d_in[23];

    // workspace layout (~147 MB)
    char* p = (char*)d_ws;
    float*          xf    = (float*)p;          p += (size_t)MR * Dc * 4;
    unsigned short* xb    = (unsigned short*)p; p += (size_t)MR * Dc * 2;
    float*          yf    = (float*)p;          p += (size_t)MR * Dc * 4;
    unsigned short* qkvb  = (unsigned short*)p; p += (size_t)MR * 3 * Dc * 2;
    unsigned short* attnb = (unsigned short*)p; p += (size_t)MR * Dc * 2;
    unsigned short* hb    = (unsigned short*)p; p += (size_t)MR * Fc * 2;
    unsigned short* latb  = (unsigned short*)p; p += (size_t)Bc * Dc * 2;
    unsigned short* vbb   = (unsigned short*)p; p += (size_t)Bc * Dc * 2;
    float*          caf   = (float*)p;          p += (size_t)Bc * Dc * 4;

    const dim3 blk(256);
    const int MT = MR / 128;  // 64 row tiles

    k_embed<<<MR, blk, 0, stream>>>(target, tok_emb, pos_emb, xf, xb);
    k_cvt<<<(Bc * Dc + 255) / 256, blk, 0, stream>>>(latent, latb, Bc * Dc);

    for (int i = 0; i < Lc; ++i) {
        // ---- self-attention ----
        k_gemm<<<dim3(3 * Dc / 128, MT), blk, 0, stream>>>(
            xb, sa_w_qkv + (size_t)i * 3 * Dc * Dc, sa_b_qkv + (size_t)i * 3 * Dc,
            nullptr, qkvb, MR, 3 * Dc, Dc, 0);
        k_attn<<<dim3(Sc / 128, Hc, Bc), blk, 0, stream>>>(qkvb, attnb);
        k_gemm<<<dim3(Dc / 128, MT), blk, 0, stream>>>(
            attnb, sa_w_o + (size_t)i * Dc * Dc, sa_b_o + (size_t)i * Dc,
            yf, nullptr, MR, Dc, Dc, 0);
        k_addln<<<MR, blk, 0, stream>>>(xf, yf, ln1_g + (size_t)i * Dc,
                                        ln1_b + (size_t)i * Dc, xf, xb, 0);

        // ---- cross-attention (exact: uniform attn over identical keys) ----
        k_gemm<<<dim3(Dc / 128, 1), blk, 0, stream>>>(
            latb,
            ca_w_qkv + (size_t)i * 3 * Dc * Dc + (size_t)2 * Dc * Dc,  // Wv
            ca_b_qkv + (size_t)i * 3 * Dc + 2 * Dc,                    // bv
            nullptr, vbb, Bc, Dc, Dc, 0);
        k_gemm<<<dim3(Dc / 128, 1), blk, 0, stream>>>(
            vbb, ca_w_o + (size_t)i * Dc * Dc, ca_b_o + (size_t)i * Dc,
            caf, nullptr, Bc, Dc, Dc, 0);
        k_addln<<<MR, blk, 0, stream>>>(xf, caf, ln2_g + (size_t)i * Dc,
                                        ln2_b + (size_t)i * Dc, xf, xb, 1);

        // ---- FFN ----
        k_gemm<<<dim3(Fc / 128, MT), blk, 0, stream>>>(
            xb, ffn_w1 + (size_t)i * Fc * Dc, ffn_b1 + (size_t)i * Fc,
            nullptr, hb, MR, Fc, Dc, 1);
        k_gemm<<<dim3(Dc / 128, MT), blk, 0, stream>>>(
            hb, ffn_w2 + (size_t)i * Dc * Fc, ffn_b2 + (size_t)i * Dc,
            yf, nullptr, MR, Dc, Fc, 0);
        k_addln<<<MR, blk, 0, stream>>>(xf, yf, ln3_g + (size_t)i * Dc,
                                        ln3_b + (size_t)i * Dc, xf, xb, 0);
    }

    // ---- logits: [MR,768] @ [100,768]^T -> d_out f32 ----
    k_gemm<<<dim3(1, MT), blk, 0, stream>>>(xb, out_w, out_b,
                                            (float*)d_out, nullptr, MR, Vc, Dc, 0);
}
